// MoELayer_33380485824726
// MI455X (gfx1250) — matrix operand order repaired and checked by reading, not compiled, not run on hardware
//
#include <hip/hip_runtime.h>

// ---------------- problem constants ----------------
#define TD   4096          // tokens (2*2048)
#define HD   1024          // hidden
#define ID   4096          // expert intermediate
#define ED   8             // routed experts
#define TOPK 2
#define MT   32            // tokens per block (2 WMMA M-tiles)
#define ICH  256           // I-chunk per stage-1/stage-2 round
#define OUTN (TD*HD)       // output elements before the loss scalar
#define MAX_TILES ((TD*TOPK)/MT + ED + TD/MT)   // 256 + 8 + 128 = 392

typedef __attribute__((ext_vector_type(16))) __bf16 v16bf;
typedef __attribute__((ext_vector_type(8)))  float  v8f;

// ---------------- helpers ----------------
static __device__ inline unsigned short bf_bits(float f) {
    unsigned u = __builtin_bit_cast(unsigned, f);
    u += 0x7FFFu + ((u >> 16) & 1u);        // round-to-nearest-even
    return (unsigned short)(u >> 16);
}

struct B256 { uint4 lo; uint4 hi; };

// A-fragment (16x32 bf16, row-major source):
// a[i] = row[k0 + 16*(i>>3) + 8*half + (i&7)] -> chunks at k0+8h and k0+16+8h.
static __device__ inline v16bf lda16(const unsigned short* row, int k0, int half) {
    B256 t;
    t.lo = *(const uint4*)(row + k0 + 8 * half);
    t.hi = *(const uint4*)(row + k0 + 16 + 8 * half);
    return __builtin_bit_cast(v16bf, t);
}

// B-fragment (32x16 bf16) from N-major (transposed) weights WT[n][k], ld=ldk:
// b[i] = WT[col][k0 + 16*(i>>3) + 8*half + (i&7)] -> chunks at k0+8h and k0+16+8h.
static __device__ inline v16bf ldb16(const unsigned short* __restrict__ WT,
                                     size_t ldk, int col, int k0, int half) {
    const unsigned short* p = WT + (size_t)col * ldk + k0 + 8 * half;
    B256 t;
    t.lo = *(const uint4*)(p);
    t.hi = *(const uint4*)(p + 16);
    return __builtin_bit_cast(v16bf, t);
}

static __device__ inline v8f wmma_bf16(v16bf a, v16bf b, v8f c) {
    return __builtin_amdgcn_wmma_f32_16x16x32_bf16(
        false, a, false, b, (short)0, c, false, false);
}
static __device__ inline v8f zero8() {
    v8f z;
#pragma unroll
    for (int i = 0; i < 8; ++i) z[i] = 0.0f;
    return z;
}
// silu(g)*u with fast v_rcp_f32 (result is rounded to bf16 anyway).
static __device__ inline float swiglu(float g, float u) {
    return g * u * __builtin_amdgcn_rcpf(1.0f + __expf(-g));
}

// ---------------- kernels ----------------
__global__ void init_ws_k(float* out, int n, int* counts, int* fill, float* zsum) {
    int i = blockIdx.x * blockDim.x + threadIdx.x;
    for (int idx = i; idx < n; idx += gridDim.x * blockDim.x) out[idx] = 0.0f;
    if (i < ED) { counts[i] = 0; fill[i] = 0; }
    if (i == 0) zsum[0] = 0.0f;
}

__global__ void xcast_k(const float* __restrict__ x, unsigned short* __restrict__ xb, int n) {
    int i = blockIdx.x * blockDim.x + threadIdx.x;
    for (int idx = i; idx < n; idx += gridDim.x * blockDim.x) xb[idx] = bf_bits(x[idx]);
}

// Transpose + cast: in [z][R][C] fp32 -> out [z][C][R] bf16.
__global__ void tcast_k(const float* __restrict__ in, unsigned short* __restrict__ out,
                        int R, int C) {
    __shared__ float tile[32][33];
    size_t base = (size_t)blockIdx.z * R * C;
    int bx = blockIdx.x * 32, by = blockIdx.y * 32;
    int tx = threadIdx.x, ty = threadIdx.y;          // 32 x 8
#pragma unroll
    for (int j = 0; j < 32; j += 8)
        tile[ty + j][tx] = in[base + (size_t)(by + ty + j) * C + bx + tx];
    __syncthreads();
#pragma unroll
    for (int j = 0; j < 32; j += 8)
        out[base + (size_t)(bx + ty + j) * R + by + tx] = bf_bits(tile[tx][ty + j]);
}

__global__ void router_k(const float* __restrict__ x, const float* __restrict__ rw,
                         int* __restrict__ topi, float* __restrict__ topw,
                         int* __restrict__ counts, float* __restrict__ zsum) {
    int t = blockIdx.x * blockDim.x + threadIdx.x;
    if (t >= TD) return;
    float lg[ED];
#pragma unroll
    for (int e = 0; e < ED; ++e) lg[e] = 0.0f;
    const float* xr = x + (size_t)t * HD;
    for (int h = 0; h < HD; ++h) {
        float xv = xr[h];
#pragma unroll
        for (int e = 0; e < ED; ++e) lg[e] += xv * rw[h * ED + e];
    }
    float z = 0.0f;
#pragma unroll
    for (int e = 0; e < ED; ++e) z += lg[e] * lg[e];
    atomicAdd(zsum, z);

    float mx = lg[0];
#pragma unroll
    for (int e = 1; e < ED; ++e) mx = fmaxf(mx, lg[e]);
    float p[ED], s = 0.0f;
#pragma unroll
    for (int e = 0; e < ED; ++e) { p[e] = __expf(lg[e] - mx); s += p[e]; }
#pragma unroll
    for (int e = 0; e < ED; ++e) p[e] /= s;

    int i0 = 0; float p0 = p[0];
#pragma unroll
    for (int e = 1; e < ED; ++e) if (p[e] > p0) { p0 = p[e]; i0 = e; }
    int i1 = (i0 == 0) ? 1 : 0; float p1 = p[i1];
#pragma unroll
    for (int e = 0; e < ED; ++e) if (e != i0 && p[e] > p1) { p1 = p[e]; i1 = e; }

    float sw = p0 + p1;
    topi[t * 2 + 0] = i0; topi[t * 2 + 1] = i1;
    topw[t * 2 + 0] = p0 / sw; topw[t * 2 + 1] = p1 / sw;
    atomicAdd(&counts[i0], 1);
    atomicAdd(&counts[i1], 1);
}

__global__ void prefix_tiles_k(const int* __restrict__ counts, int* __restrict__ offsets,
                               int* __restrict__ tile_e, int* __restrict__ tile_b,
                               int* __restrict__ tile_n) {
    int idx = 0;
    offsets[0] = 0;
    for (int e = 0; e < ED; ++e) {
        int c = counts[e];
        offsets[e + 1] = offsets[e] + c;
        int nt = (c + MT - 1) / MT;
        for (int j = 0; j < nt; ++j) {
            int rem = c - MT * j;
            tile_e[idx] = e;
            tile_b[idx] = offsets[e] + MT * j;
            tile_n[idx] = rem < MT ? rem : MT;
            ++idx;
        }
    }
    for (int j = 0; j < TD / MT; ++j) {  // shared-expert tiles
        tile_e[idx] = ED; tile_b[idx] = MT * j; tile_n[idx] = MT; ++idx;
    }
    for (; idx < MAX_TILES; ++idx) tile_e[idx] = -1;
}

__global__ void scatter_k(const int* __restrict__ topi, const float* __restrict__ topw,
                          const int* __restrict__ offsets, int* __restrict__ fill,
                          int* __restrict__ perm_tok, float* __restrict__ perm_w) {
    int t = blockIdx.x * blockDim.x + threadIdx.x;
    if (t >= TD) return;
#pragma unroll
    for (int k = 0; k < TOPK; ++k) {
        int e = topi[t * 2 + k];
        int pos = atomicAdd(&fill[e], 1);
        int slot = offsets[e] + pos;
        perm_tok[slot] = t;
        perm_w[slot] = topw[t * 2 + k];
    }
}

// Fused SwiGLU expert FFN for one (expert, 32-token) tile. 256 thr = 8 waves.
__global__ __launch_bounds__(256) void moe_ffn_k(
    const unsigned short* __restrict__ xb,
    const unsigned short* __restrict__ wgT, const unsigned short* __restrict__ wuT,
    const unsigned short* __restrict__ wdT,
    const unsigned short* __restrict__ sgT, const unsigned short* __restrict__ suT,
    const unsigned short* __restrict__ sdT,
    const int* __restrict__ tile_e, const int* __restrict__ tile_b, const int* __restrict__ tile_n,
    const int* __restrict__ perm_tok, const float* __restrict__ perm_w,
    float* __restrict__ out) {

    __shared__ unsigned short hbuf[MT * ICH];   // 16 KB SwiGLU chunk (bf16)
    __shared__ int   stok[MT];
    __shared__ float swgt[MT];

    const int bid = blockIdx.x;
    const int e = tile_e[bid];
    if (e < 0) return;

    const int tid  = threadIdx.x;
    const int lane = tid & 31;
    const int wv   = tid >> 5;
    const int half = (lane >> 4) & 1;
    const int nlo  = lane & 15;

    if (tid < MT) {
        int tok; float w;
        if (e == ED) { tok = tile_b[bid] + tid; w = 1.0f; }
        else {
            int nv = tile_n[bid];
            if (tid < nv) { int slot = tile_b[bid] + tid; tok = perm_tok[slot]; w = perm_w[slot]; }
            else          { tok = 0; w = 0.0f; }
        }
        stok[tid] = tok; swgt[tid] = w;
    }
    __syncthreads();

    const unsigned short* WG = (e == ED) ? sgT : wgT + (size_t)e * HD * ID;
    const unsigned short* WU = (e == ED) ? suT : wuT + (size_t)e * HD * ID;
    const unsigned short* WD = (e == ED) ? sdT : wdT + (size_t)e * ID * HD;

    // A rows for this lane (rows m and m+16), straight from global bf16 xb.
    const unsigned short* xrow0 = xb + (size_t)stok[nlo] * HD;
    const unsigned short* xrow1 = xb + (size_t)stok[16 + nlo] * HD;

    v8f oacc[16];   // persistent down-proj accumulators: [n-tile 0..7][m-tile 0..1]
#pragma unroll
    for (int i = 0; i < 16; ++i) oacc[i] = zero8();

    for (int ch = 0; ch < ID / ICH; ++ch) {
        // ---- stage 1: h = silu(x@Wg) * (x@Wu) for a 256-wide I chunk --------
        v8f ga[4], ua[4];   // [n-tile 0..1][m-tile 0..1]
#pragma unroll
        for (int i = 0; i < 4; ++i) { ga[i] = zero8(); ua[i] = zero8(); }

        const int cb1 = ch * ICH + wv * 32 + nlo;   // this wave's stage-1 col base
#pragma unroll 1
        for (int k0 = 0; k0 < HD; k0 += 32) {
            v16bf a0 = lda16(xrow0, k0, half);
            v16bf a1 = lda16(xrow1, k0, half);
            // Fragment sequence f0..f3 = bg0,bu0,bg1,bu1 ; 2-deep rotation so
            // f(i+2)'s loads are issued before the WMMAs consuming f(i):
            // the scheduler can wait on loadcnt<=2 instead of 0.
            v16bf bA = ldb16(WG, HD, cb1 + 0 * 16, k0, half);   // f0 = bg0
            v16bf bB = ldb16(WU, HD, cb1 + 0 * 16, k0, half);   // f1 = bu0
#pragma unroll
            for (int f = 0; f < 4; ++f) {
                int t  = f >> 1;           // n-tile of current fragment
                int gu = f & 1;            // 0 = gate, 1 = up
                v16bf bn;
                if (f + 2 < 4) {
                    bn = (f + 2 == 2) ? ldb16(WG, HD, cb1 + 16, k0, half)
                                      : ldb16(WU, HD, cb1 + 16, k0, half);
                }
                if (gu == 0) {
                    ga[t * 2 + 0] = wmma_bf16(a0, bA, ga[t * 2 + 0]);
                    ga[t * 2 + 1] = wmma_bf16(a1, bA, ga[t * 2 + 1]);
                } else {
                    ua[t * 2 + 0] = wmma_bf16(a0, bA, ua[t * 2 + 0]);
                    ua[t * 2 + 1] = wmma_bf16(a1, bA, ua[t * 2 + 1]);
                }
                bA = bB; bB = bn;
            }
        }
#pragma unroll
        for (int t = 0; t < 2; ++t) {
            int lcol = wv * 32 + t * 16 + nlo;
#pragma unroll
            for (int mt = 0; mt < 2; ++mt) {
#pragma unroll
                for (int r = 0; r < 8; ++r) {
                    int m = 16 * mt + 8 * half + r;
                    hbuf[m * ICH + lcol] = bf_bits(swiglu(ga[t * 2 + mt][r],
                                                          ua[t * 2 + mt][r]));
                }
            }
        }
        __syncthreads();

        // ---- stage 2: out += h_chunk @ Wd[chunk rows] -----------------------
        const int cb2 = wv * 128 + nlo;   // this wave's stage-2 col base
#pragma unroll 1
        for (int kk = 0; kk < ICH; kk += 32) {
            v16bf a0 = lda16(&hbuf[nlo * ICH], kk, half);
            v16bf a1 = lda16(&hbuf[(16 + nlo) * ICH], kk, half);
            int kg = ch * ICH + kk;
            // 2-deep rotating pipeline over the 8 Wd fragments.
            v16bf bA = ldb16(WD, ID, cb2 + 0 * 16, kg, half);
            v16bf bB = ldb16(WD, ID, cb2 + 1 * 16, kg, half);
#pragma unroll
            for (int t2 = 0; t2 < 8; ++t2) {
                v16bf bn;
                if (t2 + 2 < 8)
                    bn = ldb16(WD, ID, cb2 + (t2 + 2) * 16, kg, half);
                oacc[t2 * 2 + 0] = wmma_bf16(a0, bA, oacc[t2 * 2 + 0]);
                oacc[t2 * 2 + 1] = wmma_bf16(a1, bA, oacc[t2 * 2 + 1]);
                bA = bB; bB = bn;
            }
        }
        __syncthreads();   // hbuf reused next chunk
    }

    // ---- epilogue: scaled scatter-add into out ------------------------------
#pragma unroll
    for (int t2 = 0; t2 < 8; ++t2) {
        int col = wv * 128 + t2 * 16 + nlo;
#pragma unroll
        for (int mt = 0; mt < 2; ++mt) {
#pragma unroll
            for (int r = 0; r < 8; ++r) {
                int m = 16 * mt + 8 * half + r;
                float v = oacc[t2 * 2 + mt][r] * swgt[m];
                atomicAdd(&out[(size_t)stok[m] * HD + col], v);
            }
        }
    }
}

__global__ void loss_k(const int* __restrict__ counts, const float* __restrict__ zsum,
                       float* __restrict__ out) {
    float loads[ED + 1];
    float sum = 0.0f;
    for (int e = 0; e < ED; ++e) { loads[e] = (float)counts[e]; sum += loads[e]; }
    loads[ED] = (float)TD; sum += loads[ED];
    const float ideal = 1.0f / (float)(ED + 1);
    float mse = 0.0f;
    for (int e = 0; e < ED + 1; ++e) {
        float d = loads[e] / sum - ideal;
        mse += d * d;
    }
    mse /= (float)(ED + 1);
    out[OUTN] = 0.01f * mse + 0.01f * (zsum[0] / (float)TD);
}

// ---------------- launch ----------------
extern "C" void kernel_launch(void* const* d_in, const int* in_sizes, int n_in,
                              void* d_out, int out_size, void* d_ws, size_t ws_size,
                              hipStream_t stream) {
    (void)in_sizes; (void)n_in; (void)out_size; (void)ws_size;
    const float* x  = (const float*)d_in[0];
    const float* rw = (const float*)d_in[1];
    const float* wg = (const float*)d_in[2];
    const float* wu = (const float*)d_in[3];
    const float* wd = (const float*)d_in[4];
    const float* sg = (const float*)d_in[5];
    const float* su = (const float*)d_in[6];
    const float* sd = (const float*)d_in[7];
    float* out = (float*)d_out;

    char* ws = (char*)d_ws;
    size_t off = 0;
    auto carve = [&](size_t bytes) -> void* {
        void* r = ws + off;
        off += (bytes + 255) & ~(size_t)255;
        return r;
    };
    unsigned short* xb  = (unsigned short*)carve((size_t)TD * HD * 2);
    unsigned short* wgT = (unsigned short*)carve((size_t)ED * HD * ID * 2);
    unsigned short* wuT = (unsigned short*)carve((size_t)ED * HD * ID * 2);
    unsigned short* wdT = (unsigned short*)carve((size_t)ED * ID * HD * 2);
    unsigned short* sgT = (unsigned short*)carve((size_t)HD * ID * 2);
    unsigned short* suT = (unsigned short*)carve((size_t)HD * ID * 2);
    unsigned short* sdT = (unsigned short*)carve((size_t)ID * HD * 2);
    int*   topi     = (int*)  carve((size_t)TD * 2 * 4);
    float* topw     = (float*)carve((size_t)TD * 2 * 4);
    int*   counts   = (int*)  carve(64);
    int*   fill     = (int*)  carve(64);
    int*   offsets  = (int*)  carve(64);
    float* zsum     = (float*)carve(64);
    int*   perm_tok = (int*)  carve((size_t)TD * 2 * 4);
    float* perm_w   = (float*)carve((size_t)TD * 2 * 4);
    int*   tile_e   = (int*)  carve((size_t)MAX_TILES * 4);
    int*   tile_b   = (int*)  carve((size_t)MAX_TILES * 4);
    int*   tile_n   = (int*)  carve((size_t)MAX_TILES * 4);

    dim3 tb(32, 8);
    init_ws_k<<<2048, 256, 0, stream>>>(out, OUTN + 1, counts, fill, zsum);
    xcast_k<<<2048, 256, 0, stream>>>(x, xb, TD * HD);
    // Routed weights: [E][H][I] -> [E][I][H] bf16 (and [E][I][H] -> [E][H][I])
    tcast_k<<<dim3(ID / 32, HD / 32, ED), tb, 0, stream>>>(wg, wgT, HD, ID);
    tcast_k<<<dim3(ID / 32, HD / 32, ED), tb, 0, stream>>>(wu, wuT, HD, ID);
    tcast_k<<<dim3(HD / 32, ID / 32, ED), tb, 0, stream>>>(wd, wdT, ID, HD);
    tcast_k<<<dim3(ID / 32, HD / 32, 1), tb, 0, stream>>>(sg, sgT, HD, ID);
    tcast_k<<<dim3(ID / 32, HD / 32, 1), tb, 0, stream>>>(su, suT, HD, ID);
    tcast_k<<<dim3(HD / 32, ID / 32, 1), tb, 0, stream>>>(sd, sdT, ID, HD);

    router_k<<<TD / 256, 256, 0, stream>>>(x, rw, topi, topw, counts, zsum);
    prefix_tiles_k<<<1, 1, 0, stream>>>(counts, offsets, tile_e, tile_b, tile_n);
    scatter_k<<<TD / 256, 256, 0, stream>>>(topi, topw, offsets, fill, perm_tok, perm_w);
    moe_ffn_k<<<MAX_TILES, 256, 0, stream>>>(xb, wgT, wuT, wdT, sgT, suT, sdT,
                                             tile_e, tile_b, tile_n,
                                             perm_tok, perm_w, out);
    loss_k<<<1, 1, 0, stream>>>(counts, zsum, out);
}
